// BarycentricPooling_901943132315
// MI455X (gfx1250) — matrix operand order repaired and checked by reading, not compiled, not run on hardware
//
#include <hip/hip_runtime.h>
#include <stdint.h>

#define NUM_GRAPHS 64
#define CODEBOOK   64
#define HID        256
#define DIST       8
#define N_NODES    50000
#define P_ROWS     (N_NODES * DIST)   /* 400000, = 3125 * 128 exactly */
#define EPS        0.1f
#define ITERS      20
#define STABF      1e-8f
#define CLAMP_MAXF 1e6f

typedef __attribute__((ext_vector_type(16))) __bf16        v16bf;
typedef __attribute__((ext_vector_type(8)))  float         v8f;
typedef __attribute__((ext_vector_type(8)))  unsigned int  v8u;

// ---------------------------------------------------------------------------
// f32 -> bf16 (RNE, used only for the tiny codebook conversion)
// ---------------------------------------------------------------------------
__device__ __forceinline__ unsigned int bf16_bits_rne(float f) {
    unsigned int u = __float_as_uint(f);
    return (u + 0x7FFFu + ((u >> 16) & 1u)) >> 16;
}
// Hot-path pack: one v_perm_b32 builds {hi.bf16, lo.bf16} (truncation).
__device__ __forceinline__ unsigned int pack2_bf16(float lo, float hi) {
    return __builtin_amdgcn_perm(__float_as_uint(hi), __float_as_uint(lo),
                                 0x07060302u);
}

// ---------------------------------------------------------------------------
// Kernel 1: zero per-graph counts, codebook row norms, codebook f32 -> bf16
// ---------------------------------------------------------------------------
__global__ void bp_init_kernel(const float* __restrict__ cb,
                               unsigned short* __restrict__ cb_bf,
                               float* __restrict__ ynorm,
                               int* __restrict__ counts) {
    int tid = threadIdx.x;
    for (int i = tid; i < CODEBOOK * HID; i += 256)
        cb_bf[i] = (unsigned short)bf16_bits_rne(cb[i]);
    if (tid < NUM_GRAPHS) counts[tid] = 0;
    if (tid < CODEBOOK) {
        float s = 0.f;
        const float* row = cb + tid * HID;
        for (int j = 0; j < HID; ++j) s += row[j] * row[j];
        ynorm[tid] = s;
    }
}

// ---------------------------------------------------------------------------
// Kernel 2: histogram of batch_idx (integer atomics -> deterministic)
// ---------------------------------------------------------------------------
__global__ void bp_count_kernel(const int* __restrict__ batch,
                                int* __restrict__ counts) {
    int i = blockIdx.x * blockDim.x + threadIdx.x;
    if (i < N_NODES) atomicAdd(&counts[batch[i]], 1);
}

// ---------------------------------------------------------------------------
// Kernel 3: exclusive scan of 64 counts (trivial, single lane)
// ---------------------------------------------------------------------------
__global__ void bp_scan_kernel(const int* __restrict__ counts,
                               int* __restrict__ rowstart) {
    if (threadIdx.x == 0) {
        int acc = 0;
        rowstart[0] = 0;
        for (int b = 0; b < NUM_GRAPHS; ++b) {
            acc += counts[b];
            rowstart[b + 1] = acc;
        }
    }
}

// ---------------------------------------------------------------------------
// Kernel 4: Kmat = exp(-max(|x|^2 + |c|^2 - 2 x.cT, 0)/eps) via bf16 WMMA.
// Block = 256 threads = 8 waves; wave owns a 16-row x 64-col output tile.
// ---------------------------------------------------------------------------
__global__ __launch_bounds__(256) void bp_gemm_kernel(
        const float* __restrict__ X,
        const unsigned short* __restrict__ cb_bf,
        const float* __restrict__ ynorm,
        float* __restrict__ Kmat) {
    __shared__ __align__(32) unsigned short sCB[CODEBOOK * HID];  // 32 KB bf16
    __shared__ float sYN[CODEBOOK];
    __shared__ float sXN[128];

    const int tid = threadIdx.x;

    // Stage codebook (bf16) into LDS: 2048 uint4's across 256 threads.
    {
        const uint4* src = (const uint4*)cb_bf;
        uint4* dst = (uint4*)sCB;
        #pragma unroll
        for (int i = 0; i < 8; ++i)
            dst[tid + i * 256] = src[tid + i * 256];
        if (tid < CODEBOOK) sYN[tid] = ynorm[tid];
    }
    __syncthreads();

    const int lane   = tid & 31;
    const int w      = tid >> 5;
    const int laneHi = lane >> 4;      // 0: K {0..15}; 1: K {16..31}
    const int m      = lane & 15;      // row within wave tile / col within n-tile

    const long long rowBase = (long long)blockIdx.x * 128 + (long long)w * 16;
    const float* rowPtr = X + (rowBase + m) * (long long)HID;

    v8f c[4];
    {
        v8f z = {0.f, 0.f, 0.f, 0.f, 0.f, 0.f, 0.f, 0.f};
        c[0] = z; c[1] = z; c[2] = z; c[3] = z;
    }

    float xnp = 0.f;                   // per-lane partial |x|^2

    #pragma unroll
    for (int ks = 0; ks < 8; ++ks) {   // K = 256 in steps of 32
        const int k0 = ks * 32 + laneHi * 16;
        if (ks < 7) __builtin_prefetch(rowPtr + k0 + 32, 0, 1);

        float4 f0 = *(const float4*)(rowPtr + k0);
        float4 f1 = *(const float4*)(rowPtr + k0 + 4);
        float4 f2 = *(const float4*)(rowPtr + k0 + 8);
        float4 f3 = *(const float4*)(rowPtr + k0 + 12);

        xnp += f0.x*f0.x + f0.y*f0.y + f0.z*f0.z + f0.w*f0.w
             + f1.x*f1.x + f1.y*f1.y + f1.z*f1.z + f1.w*f1.w
             + f2.x*f2.x + f2.y*f2.y + f2.z*f2.z + f2.w*f2.w
             + f3.x*f3.x + f3.y*f3.y + f3.z*f3.z + f3.w*f3.w;

        // A fragment: lane holds 16 contiguous K's (matches B order);
        // one v_perm_b32 per packed bf16 pair.
        v8u ap;
        ap[0] = pack2_bf16(f0.x, f0.y);  ap[1] = pack2_bf16(f0.z, f0.w);
        ap[2] = pack2_bf16(f1.x, f1.y);  ap[3] = pack2_bf16(f1.z, f1.w);
        ap[4] = pack2_bf16(f2.x, f2.y);  ap[5] = pack2_bf16(f2.z, f2.w);
        ap[6] = pack2_bf16(f3.x, f3.y);  ap[7] = pack2_bf16(f3.z, f3.w);
        v16bf a = __builtin_bit_cast(v16bf, ap);

        // B fragment: lane holds 16 contiguous K's of codebook row (n*16+m)
        const int kk = ks * 32 + laneHi * 16;
        #pragma unroll
        for (int n = 0; n < 4; ++n) {
            v16bf b = __builtin_bit_cast(
                v16bf, *(const v8u*)&sCB[(n * 16 + m) * HID + kk]);
            c[n] = __builtin_amdgcn_wmma_f32_16x16x32_bf16(
                false, a, false, b, (short)0, c[n], false, false);
        }
    }

    // Complete |x|^2 per row: partner lane holds the other K-chunks.
    float xn = xnp + __shfl_xor(xnp, 16, 32);
    if (laneHi == 0) sXN[w * 16 + m] = xn;
    __syncthreads();

    // Epilogue: cost -> exp -> NaN/Inf guard -> Kmat (C layout: M=r+8*hi, N=m)
    #pragma unroll
    for (int n = 0; n < 4; ++n) {
        const int col = n * 16 + m;
        const float yn = sYN[col];
        #pragma unroll
        for (int r = 0; r < 8; ++r) {
            const int rr = r + laneHi * 8;
            float cost = fmaxf(sXN[w * 16 + rr] + yn - 2.0f * c[n][r], 0.f);
            float kv = __expf(-cost * (1.0f / EPS));
            if (!__builtin_isfinite(kv)) kv = STABF;
            Kmat[(rowBase + rr) * 64 + col] = kv;
        }
    }
}

// ---------------------------------------------------------------------------
// Kernel 5: fused Sinkhorn. One block per graph (contiguous row range),
// 32 waves, tiles of 32 rows per wave:
//   pass 1 (lane-owns-row):   Kv row-dot with no cross-lane reduction,
//                             u = a * rcp(max(Kv, eps)), staged to LDS.
//   pass 2 (lane-owns-2cols): KTu accumulation, u broadcast from LDS.
// Deterministic fixed-order LDS serial reduction across waves per iteration.
// Final wsum accumulation fused into iteration 20 (u20 = f(v19)).
// ---------------------------------------------------------------------------
__global__ __launch_bounds__(1024) void bp_sinkhorn_kernel(
        const float* __restrict__ Kmat,
        const int* __restrict__ counts,
        const int* __restrict__ rowstart,
        float* __restrict__ out) {
    __shared__ __align__(16) float sV[CODEBOOK];
    __shared__ __align__(16) float sU[32 * 32];   // per-wave u staging
    __shared__ float sW[CODEBOOK];
    __shared__ float sPartK[32 * 65];   // stride 65: conflict-free column sums
    __shared__ float sPartW[32 * 65];
    __shared__ float sTot;

    const int b   = blockIdx.x;
    const int tid = threadIdx.x;
    const int cnt = counts[b];

    if (cnt == 0) {                     // empty graph -> zeros (reference)
        if (tid < CODEBOOK) out[b * CODEBOOK + tid] = 0.f;
        return;
    }

    const int   pstart = rowstart[b] * DIST;
    const int   pend   = rowstart[b + 1] * DIST;
    const float a      = 1.0f / (float)(cnt * DIST);

    if (tid < CODEBOOK) sV[tid] = 1.0f;
    __syncthreads();

    const int lane = tid & 31;
    const int w    = tid >> 5;

    for (int t = 0; t < ITERS; ++t) {
        const bool last = (t == ITERS - 1);
        float k0a = 0.f, k1a = 0.f, w0a = 0.f, w1a = 0.f;

        for (int tb = pstart + w * 32; tb < pend; tb += 32 * 32) {
            // ---- pass 1: each lane computes Kv for its own row ----
            const int p = tb + lane;
            float kv = 0.f;
            if (p < pend) {
                const float4* rp = (const float4*)(Kmat + (long long)p * 64);
                #pragma unroll
                for (int j = 0; j < 16; ++j) {
                    float4 kq = rp[j];
                    float4 vq = *(const float4*)&sV[4 * j];   // LDS broadcast
                    kv += kq.x*vq.x + kq.y*vq.y + kq.z*vq.z + kq.w*vq.w;
                }
            }
            float u = a * __builtin_amdgcn_rcpf(fmaxf(kv, STABF));
            sU[w * 32 + lane] = u;       // same-wave LDS: no barrier needed

            // ---- pass 2: each lane accumulates KTu for its 2 columns ----
            const int rows = min(32, pend - tb);
            for (int r4 = 0; r4 < rows; r4 += 4) {
                float4 uq = *(const float4*)&sU[w * 32 + r4];  // broadcast
                float us[4] = {uq.x, uq.y, uq.z, uq.w};
                #pragma unroll
                for (int r = 0; r < 4; ++r) {
                    if (r4 + r >= rows) break;
                    float2 kk = *(const float2*)(
                        Kmat + (long long)(tb + r4 + r) * 64 + 2 * lane);
                    float u_ = us[r];
                    k0a += kk.x * u_;  k1a += kk.y * u_;
                    if (last) {
                        float uc = fminf(fmaxf(u_, STABF), CLAMP_MAXF);
                        w0a += kk.x * uc;  w1a += kk.y * uc;
                    }
                }
            }
        }

        sPartK[w * 65 + 2 * lane]     = k0a;
        sPartK[w * 65 + 2 * lane + 1] = k1a;
        if (last) {
            sPartW[w * 65 + 2 * lane]     = w0a;
            sPartW[w * 65 + 2 * lane + 1] = w1a;
        }
        __syncthreads();

        if (tid < CODEBOOK) {
            float s = 0.f;
            #pragma unroll
            for (int ww = 0; ww < 32; ++ww) s += sPartK[ww * 65 + tid];
            sV[tid] = (1.0f / (float)CODEBOOK) / fmaxf(s, STABF);
            if (last) {
                float sw = 0.f;
                #pragma unroll
                for (int ww = 0; ww < 32; ++ww) sw += sPartW[ww * 65 + tid];
                sW[tid] = sw;
            }
        }
        __syncthreads();
    }

    if (tid < CODEBOOK) {
        float vc = fminf(fmaxf(sV[tid], STABF), CLAMP_MAXF);
        sW[tid] = sW[tid] * vc;                          // wsum[b,k]
    }
    __syncthreads();
    if (tid == 0) {
        float tot = 0.f;
        for (int k = 0; k < CODEBOOK; ++k) tot += sW[k]; // fixed order
        sTot = tot;
    }
    __syncthreads();
    if (tid < CODEBOOK) {
        float tot = sTot;
        out[b * CODEBOOK + tid] =
            (tot > STABF) ? sW[tid] / fmaxf(tot, STABF) : (1.0f / (float)CODEBOOK);
    }
}

// ---------------------------------------------------------------------------
extern "C" void kernel_launch(void* const* d_in, const int* in_sizes, int n_in,
                              void* d_out, int out_size, void* d_ws, size_t ws_size,
                              hipStream_t stream) {
    const float* X     = (const float*)d_in[0];  // [50000, 8, 256] f32
    const int*   batch = (const int*)d_in[1];    // [50000] int
    const float* CB    = (const float*)d_in[2];  // [64, 256] f32
    float*       out   = (float*)d_out;          // [64, 64] f32

    char* ws = (char*)d_ws;
    size_t off = 0;
    float* Kmat = (float*)(ws + off);            off += (size_t)P_ROWS * 64 * sizeof(float);
    unsigned short* cb_bf = (unsigned short*)(ws + off); off += CODEBOOK * HID * sizeof(unsigned short);
    float* ynorm = (float*)(ws + off);           off += CODEBOOK * sizeof(float);
    int* counts  = (int*)(ws + off);             off += NUM_GRAPHS * sizeof(int);
    int* rowstart = (int*)(ws + off);            off += (NUM_GRAPHS + 1) * sizeof(int);
    (void)in_sizes; (void)n_in; (void)out_size; (void)ws_size;

    bp_init_kernel<<<1, 256, 0, stream>>>(CB, cb_bf, ynorm, counts);
    bp_count_kernel<<<(N_NODES + 255) / 256, 256, 0, stream>>>(batch, counts);
    bp_scan_kernel<<<1, 32, 0, stream>>>(counts, rowstart);
    bp_gemm_kernel<<<P_ROWS / 128, 256, 0, stream>>>(X, cb_bf, ynorm, Kmat);
    bp_sinkhorn_kernel<<<NUM_GRAPHS, 1024, 0, stream>>>(Kmat, counts, rowstart, out);
}
